// MoleculeGenerator_24713241821485
// MI455X (gfx1250) — compile-verified, matrix operand order checked by reading, not run on hardware
//
#include <hip/hip_runtime.h>
#include <hip/hip_bf16.h>

// ---------------------------------------------------------------------------
// MoleculeGenerator decoder for MI455X (gfx1250, wave32, WMMA bf16).
// L=3, H=V=512, B=16, S=96.
// All GEMM operands (A and B) are pre-packed into bf16 WMMA tile layout so
// the serial-scan inner loops are pure {2x b128 load A, 2x b128 load B, wmma}.
// f32 master copies of the recurrent state are kept for accuracy.
// ---------------------------------------------------------------------------

#define GH 512          // hidden dim
#define GB 16           // batch
#define GS 96           // sequence length
#define G3H 1536        // 3*H
#define KB512 16        // K-blocks for K=512

typedef __attribute__((ext_vector_type(16))) __bf16 v16bf;
typedef __attribute__((ext_vector_type(8)))  float  v8f;

// D = A(16x32 bf16) * B(32x16 bf16) + C(16x16 f32)
__device__ __forceinline__ v8f wmma_bf16(v16bf a, v16bf b, v8f c) {
    return __builtin_amdgcn_wmma_f32_16x16x32_bf16(
        /*neg_a=*/false, a, /*neg_b=*/false, b,
        /*c_mod=*/(short)0, c, /*reuse_a=*/false, /*reuse_b=*/false);
}

// Packed tile load (A or B operand): lane's 16 bf16 (32 bytes) contiguous.
// T[((tb*KB + kb)*32 + lane)*16 + m]
__device__ __forceinline__ v16bf load_tile(const __bf16* __restrict__ T,
                                           int KB, int tb, int kb, int lane) {
    return *(const v16bf*)(T + (((size_t)(tb * KB + kb)) * 32 + lane) * 16);
}

// Scatter one f32 value into packed-A bf16 layout at (global row R, col k).
// A layout: lane = (R&15) + 16*khalf, element m with
//   k = kb*32 + (m>>3)*16 + khalf*8 + (m&7)
__device__ __forceinline__ void pack_store(__bf16* __restrict__ P, int KB,
                                           int R, int k, float v) {
    int mt = R >> 4, r = R & 15;
    int kb = k >> 5, rem = k & 31;
    int khalf = (rem >> 3) & 1;
    int m = ((rem >> 4) << 3) | (rem & 7);
    int lane = r + (khalf << 4);
    P[(((size_t)(mt * KB + kb)) * 32 + lane) * 16 + m] = (__bf16)v;
}

__device__ __forceinline__ float sigmoidf_(float x) {
    return 1.0f / (1.0f + expf(-x));
}

// ---------------------------------------------------------------------------
// Pre-tile a row-major f32 weight [K x N] into bf16 WMMA B-operand order.
// ---------------------------------------------------------------------------
__global__ void k_tile_w(const float* __restrict__ W, __bf16* __restrict__ T,
                         int K, int N) {
    int id = blockIdx.x * 256 + threadIdx.x;
    if (id >= K * N) return;
    int m    = id & 15;
    int lane = (id >> 4) & 31;
    int t    = id >> 9;
    int KB   = K >> 5;
    int kb   = t % KB;
    int nb   = t / KB;
    int col = lane & 15, khalf = lane >> 4;
    int k = kb * 32 + ((m >> 3) << 4) + khalf * 8 + (m & 7);
    int n = nb * 16 + col;
    T[id] = (__bf16)W[(size_t)k * N + n];
}

// ---------------------------------------------------------------------------
// Pack a row-major f32 activation [M x 512] into bf16 WMMA A-operand order.
// ---------------------------------------------------------------------------
__global__ void k_pack_a(const float* __restrict__ A, __bf16* __restrict__ P,
                         int M, int lda) {
    int id = blockIdx.x * 256 + threadIdx.x;
    if (id >= M * GH) return;
    int m    = id & 15;
    int lane = (id >> 4) & 31;
    int t    = id >> 9;
    int kb   = t % KB512;
    int mt   = t / KB512;
    int r = lane & 15, khalf = lane >> 4;
    int k = kb * 32 + ((m >> 3) << 4) + khalf * 8 + (m & 7);
    P[id] = (__bf16)A[(size_t)(mt * 16 + r) * lda + k];
}

// ---------------------------------------------------------------------------
// Generic dense: out = act(Apacked[Mx512] @ W + bias). One wave per tile.
// grid = (N/16, M/16), block = 32. Optionally emits packed-A-format output
// (for chaining GEMMs); pack K-blocks = gridDim.x/2 (N/32).
// ---------------------------------------------------------------------------
__global__ void k_dense(const __bf16* __restrict__ AP,
                        const __bf16* __restrict__ T,
                        const float* __restrict__ bias,
                        float* __restrict__ out, int ldo, int relu_flag,
                        __bf16* __restrict__ packout) {
    int lane = threadIdx.x & 31;
    int nb = blockIdx.x;
    int mt = blockIdx.y;
    v8f acc = {};
    for (int kb = 0; kb < KB512; ++kb) {
        v16bf a = load_tile(AP, KB512, mt, kb, lane);
        v16bf b = load_tile(T, KB512, nb, kb, lane);
        acc = wmma_bf16(a, b, acc);
    }
    int col = lane & 15, mhalf = lane >> 4;
    float bv = bias ? bias[nb * 16 + col] : 0.0f;
    int KBo = gridDim.x >> 1;   // output width / 32
#pragma unroll
    for (int v = 0; v < 8; ++v) {
        int r = v + 8 * mhalf;
        float val = acc[v] + bv;
        if (relu_flag) val = fmaxf(val, 0.0f);
        out[(size_t)(mt * 16 + r) * ldo + nb * 16 + col] = val;
        if (packout) pack_store(packout, KBo, mt * 16 + r, nb * 16 + col, val);
    }
}

// ---------------------------------------------------------------------------
// Fused GRU layer: wave computes gate columns j0, j0+512, j0+1024 of both
// gi and gh via WMMA, then does r/z/n nonlinearity entirely in registers.
// Reads packed bf16 A operands; writes h_new as f32 (state) + packed bf16.
// grid = 32 waves (j0 = blockIdx.x * 16), block = 32.  hout != h (dbl buffer).
// ---------------------------------------------------------------------------
template <bool HASX>
__global__ void k_gru(const float* __restrict__ h,
                      const __bf16* __restrict__ hP,
                      const __bf16* __restrict__ xP,
                      const __bf16* __restrict__ Thh,
                      const __bf16* __restrict__ Tih,
                      const float* __restrict__ bih,
                      const float* __restrict__ bhh,
                      float* __restrict__ hout,
                      __bf16* __restrict__ houtP) {
    int lane = threadIdx.x & 31;
    int tile = blockIdx.x;        // 0..31
    int nb_r = tile, nb_z = tile + 32, nb_n = tile + 64;   // NB = 96
    v8f hr = {}, hz = {}, hn = {}, ir = {}, iz = {}, in_ = {};
    for (int kb = 0; kb < KB512; ++kb) {
        v16bf ah = load_tile(hP, KB512, 0, kb, lane);
        hr = wmma_bf16(ah, load_tile(Thh, KB512, nb_r, kb, lane), hr);
        hz = wmma_bf16(ah, load_tile(Thh, KB512, nb_z, kb, lane), hz);
        hn = wmma_bf16(ah, load_tile(Thh, KB512, nb_n, kb, lane), hn);
        if constexpr (HASX) {
            v16bf ax = load_tile(xP, KB512, 0, kb, lane);
            ir  = wmma_bf16(ax, load_tile(Tih, KB512, nb_r, kb, lane), ir);
            iz  = wmma_bf16(ax, load_tile(Tih, KB512, nb_z, kb, lane), iz);
            in_ = wmma_bf16(ax, load_tile(Tih, KB512, nb_n, kb, lane), in_);
        }
    }
    int col = lane & 15, mhalf = lane >> 4;
    int j = tile * 16 + col;
    float bir = bih[j], biz = bih[512 + j], bin = bih[1024 + j];
    float bhr = bhh[j], bhz = bhh[512 + j], bhn = bhh[1024 + j];
#pragma unroll
    for (int v = 0; v < 8; ++v) {
        int r = v + 8 * mhalf;
        float gr = sigmoidf_(ir[v] + bir + hr[v] + bhr);
        float gz = sigmoidf_(iz[v] + biz + hz[v] + bhz);
        float gn = tanhf(in_[v] + bin + gr * (hn[v] + bhn));
        float ho = h[(size_t)r * GH + j];
        float hv = (1.0f - gz) * gn + gz * ho;
        hout[(size_t)r * GH + j] = hv;
        pack_store(houtP, KB512, r, j, hv);
    }
}

// ---------------------------------------------------------------------------
// argmax over V=512 logits per (b,s) row; jnp.argmax first-max tie-break.
// grid = B*S waves.
// ---------------------------------------------------------------------------
__global__ void k_argmax(const float* __restrict__ atoms, float* __restrict__ out) {
    int rowid = blockIdx.x;
    int lane  = threadIdx.x;
    const float* p = atoms + (size_t)rowid * GH;
    float best = -__builtin_inff();
    int bi = 0;
    for (int c = lane; c < GH; c += 32) {
        float v = p[c];
        if (v > best) { best = v; bi = c; }
    }
    for (int off = 16; off > 0; off >>= 1) {
        float ob = __shfl_down(best, off, 32);
        int   oi = __shfl_down(bi,   off, 32);
        if (ob > best || (ob == best && oi < bi)) { best = ob; bi = oi; }
    }
    if (lane == 0) out[rowid] = (float)bi;
}

__global__ void k_copy2(const float* __restrict__ src, float* __restrict__ d1,
                        float* __restrict__ d2, int n) {
    int i = blockIdx.x * 256 + threadIdx.x;
    if (i < n) { float v = src[i]; d1[i] = v; d2[i] = v; }
}

__global__ void k_copy2p(const __bf16* __restrict__ src, __bf16* __restrict__ d1,
                         __bf16* __restrict__ d2, int n) {
    int i = blockIdx.x * 256 + threadIdx.x;
    if (i < n) { __bf16 v = src[i]; d1[i] = v; d2[i] = v; }
}

// ---------------------------------------------------------------------------
// Fused bond MLP: edge[b,i,j,:] = relu(e_i[b,i]+e_j[b,j]+bb1) @ Wb2 + bb2,
// diagonal zeroed. Never materializes the [B,S,S,H] tensor.
// grid = (S, B), block = 256 (8 waves; wave w handles j = w, w+8, ...).
// ---------------------------------------------------------------------------
__global__ void k_edge(const float* __restrict__ e_i, const float* __restrict__ e_j,
                       const float* __restrict__ bb1, const float* __restrict__ Wb2,
                       const float* __restrict__ bb2, float* __restrict__ out) {
    __shared__ float s_ei[GH];
    __shared__ float s_w[4 * GH];     // transposed Wb2: [c][h], conflict-free
    int i = blockIdx.x, b = blockIdx.y;
    int tid = threadIdx.x;
    for (int h = tid; h < GH; h += 256)
        s_ei[h] = e_i[((size_t)(b * GS + i)) * GH + h] + bb1[h];
    for (int idx = tid; idx < 4 * GH; idx += 256) {
        int c = idx >> 9, h = idx & (GH - 1);
        s_w[idx] = Wb2[(size_t)h * 4 + c];
    }
    __syncthreads();
    int wave = tid >> 5, lane = tid & 31;
    for (int j = wave; j < GS; j += 8) {
        const float* ej = e_j + ((size_t)(b * GS + j)) * GH;
        float a0 = 0.f, a1 = 0.f, a2 = 0.f, a3 = 0.f;
        for (int h = lane; h < GH; h += 32) {
            float t = fmaxf(s_ei[h] + ej[h], 0.0f);
            a0 = fmaf(t, s_w[0 * GH + h], a0);
            a1 = fmaf(t, s_w[1 * GH + h], a1);
            a2 = fmaf(t, s_w[2 * GH + h], a2);
            a3 = fmaf(t, s_w[3 * GH + h], a3);
        }
        for (int off = 16; off > 0; off >>= 1) {
            a0 += __shfl_down(a0, off, 32);
            a1 += __shfl_down(a1, off, 32);
            a2 += __shfl_down(a2, off, 32);
            a3 += __shfl_down(a3, off, 32);
        }
        if (lane == 0) {
            size_t o = (((size_t)b * GS + i) * GS + j) * 4;
            if (i == j) {
                out[o] = 0.f; out[o + 1] = 0.f; out[o + 2] = 0.f; out[o + 3] = 0.f;
            } else {
                out[o]     = a0 + bb2[0];
                out[o + 1] = a1 + bb2[1];
                out[o + 2] = a2 + bb2[2];
                out[o + 3] = a3 + bb2[3];
            }
        }
    }
}

// ---------------------------------------------------------------------------
extern "C" void kernel_launch(void* const* d_in, const int* in_sizes, int n_in,
                              void* d_out, int out_size, void* d_ws, size_t ws_size,
                              hipStream_t stream) {
    const float* z     = (const float*)d_in[0];
    const float* W_l2h = (const float*)d_in[1];
    const float* b_l2h = (const float*)d_in[2];
    const float* W_ih  = (const float*)d_in[3];   // [3][512][1536]
    const float* W_hh  = (const float*)d_in[4];   // [3][512][1536]
    const float* b_ih  = (const float*)d_in[5];   // [3][1536]
    const float* b_hh  = (const float*)d_in[6];
    const float* Wa1   = (const float*)d_in[7];
    const float* ba1   = (const float*)d_in[8];
    const float* Wa2   = (const float*)d_in[9];
    const float* ba2   = (const float*)d_in[10];
    const float* Wb1   = (const float*)d_in[11];  // [1024][512]
    const float* bb1   = (const float*)d_in[12];
    const float* Wb2   = (const float*)d_in[13];  // [512][4]
    const float* bb2   = (const float*)d_in[14];
    (void)in_sizes; (void)n_in; (void)out_size; (void)ws_size;

    // ---- workspace carve-up (~22 MB) ----
    char* ws = (char*)d_ws;
    size_t off = 0;
    auto carve = [&](size_t bytes) -> char* {
        char* p = ws + off;
        off += (bytes + 255) & ~(size_t)255;
        return p;
    };
    const int HS = GB * GH;                       // 8192 elems per layer state
    float*  hidA  = (float*)carve(3 * HS * 4);
    float*  hidB  = (float*)carve(3 * HS * 4);
    float*  a1buf = (float*)carve(HS * 4);
    float*  atoms = (float*)carve((size_t)GB * GS * GH * 4);   // [B][S][V]
    float*  e_i   = (float*)carve((size_t)GB * GS * GH * 4);
    float*  e_j   = (float*)carve((size_t)GB * GS * GH * 4);
    __bf16* hidAP = (__bf16*)carve(3 * HS * 2);   // packed-A mirrors of state
    __bf16* hidBP = (__bf16*)carve(3 * HS * 2);
    __bf16* zP    = (__bf16*)carve(HS * 2);
    __bf16* a1P   = (__bf16*)carve(HS * 2);
    __bf16* atomsP= (__bf16*)carve((size_t)GB * GS * GH * 2);
    __bf16* T_l2h = (__bf16*)carve((size_t)GH * GH * 2);
    __bf16* T_hh0 = (__bf16*)carve((size_t)GH * G3H * 2);
    __bf16* T_hh1 = (__bf16*)carve((size_t)GH * G3H * 2);
    __bf16* T_hh2 = (__bf16*)carve((size_t)GH * G3H * 2);
    __bf16* T_ih1 = (__bf16*)carve((size_t)GH * G3H * 2);
    __bf16* T_ih2 = (__bf16*)carve((size_t)GH * G3H * 2);
    __bf16* T_a1  = (__bf16*)carve((size_t)GH * GH * 2);
    __bf16* T_a2  = (__bf16*)carve((size_t)GH * GH * 2);
    __bf16* T_b1i = (__bf16*)carve((size_t)GH * GH * 2);
    __bf16* T_b1j = (__bf16*)carve((size_t)GH * GH * 2);

    auto tileW = [&](const float* W, __bf16* T, int K, int N) {
        int total = K * N;
        k_tile_w<<<(total + 255) / 256, 256, 0, stream>>>(W, T, K, N);
    };
    // ---- one-time weight conversion + B-tile layout (bf16) ----
    tileW(W_l2h,                T_l2h, GH, GH);
    tileW(W_hh + 0 * GH * G3H,  T_hh0, GH, G3H);
    tileW(W_hh + 1 * GH * G3H,  T_hh1, GH, G3H);
    tileW(W_hh + 2 * GH * G3H,  T_hh2, GH, G3H);
    tileW(W_ih + 1 * GH * G3H,  T_ih1, GH, G3H);
    tileW(W_ih + 2 * GH * G3H,  T_ih2, GH, G3H);
    tileW(Wa1,                  T_a1,  GH, GH);
    tileW(Wa2,                  T_a2,  GH, GH);
    tileW(Wb1,                  T_b1i, GH, GH);   // Wb1[:V]
    tileW(Wb1 + GH * GH,        T_b1j, GH, GH);   // Wb1[V:]

    // ---- h0 = z @ W_l2h + b_l2h, replicated (f32 + packed) to 3 layers ----
    k_pack_a<<<(HS + 255) / 256, 256, 0, stream>>>(z, zP, GB, GH);
    k_dense<<<dim3(GH / 16, 1), 32, 0, stream>>>(zP, T_l2h, b_l2h,
                                                 hidA, GH, 0, hidAP);
    k_copy2<<<(HS + 255) / 256, 256, 0, stream>>>(hidA, hidA + HS,
                                                  hidA + 2 * HS, HS);
    k_copy2p<<<(HS + 255) / 256, 256, 0, stream>>>(hidAP, hidAP + HS,
                                                   hidAP + 2 * HS, HS);

    // ---- sequential GRU scan (double-buffered hidden state) ----
    for (int s = 0; s < GS; ++s) {
        float*  cur  = (s & 1) ? hidB  : hidA;
        float*  nxt  = (s & 1) ? hidA  : hidB;
        __bf16* curP = (s & 1) ? hidBP : hidAP;
        __bf16* nxtP = (s & 1) ? hidAP : hidBP;
        // layer 0: decoder input is zeros -> gi = b_ih[0]
        k_gru<false><<<32, 32, 0, stream>>>(cur, curP, nullptr, T_hh0, nullptr,
                                            b_ih, b_hh, nxt, nxtP);
        k_gru<true><<<32, 32, 0, stream>>>(cur + HS, curP + HS, nxtP,
                                           T_hh1, T_ih1,
                                           b_ih + G3H, b_hh + G3H,
                                           nxt + HS, nxtP + HS);
        k_gru<true><<<32, 32, 0, stream>>>(cur + 2 * HS, curP + 2 * HS,
                                           nxtP + HS, T_hh2, T_ih2,
                                           b_ih + 2 * G3H, b_hh + 2 * G3H,
                                           nxt + 2 * HS, nxtP + 2 * HS);
        // atom MLP: logits -> atoms[:, s, :]
        k_dense<<<dim3(GH / 16, 1), 32, 0, stream>>>(nxtP + 2 * HS, T_a1, ba1,
                                                     a1buf, GH, 1, a1P);
        k_dense<<<dim3(GH / 16, 1), 32, 0, stream>>>(a1P, T_a2, ba2,
                                                     atoms + (size_t)s * GH,
                                                     GS * GH, 0, nullptr);
    }

    // ---- output 0: atom indices (as float) ----
    k_argmax<<<GB * GS, 32, 0, stream>>>(atoms, (float*)d_out);

    // ---- bond MLP: pack atoms once, then e_i / e_j GEMMs ----
    {
        int total = GB * GS * GH;
        k_pack_a<<<(total + 255) / 256, 256, 0, stream>>>(atoms, atomsP,
                                                          GB * GS, GH);
    }
    k_dense<<<dim3(GH / 16, GB * GS / 16), 32, 0, stream>>>(atomsP, T_b1i,
                                                            nullptr, e_i, GH,
                                                            0, nullptr);
    k_dense<<<dim3(GH / 16, GB * GS / 16), 32, 0, stream>>>(atomsP, T_b1j,
                                                            nullptr, e_j, GH,
                                                            0, nullptr);

    // ---- output 1: fused all-pairs edge MLP ----
    k_edge<<<dim3(GS, GB), 256, 0, stream>>>(e_i, e_j, bb1, Wb2, bb2,
                                             (float*)d_out + GB * GS);
}
